// Sequence_49864570306840
// MI455X (gfx1250) — compile-verified, matrix operand order checked by reading, not run on hardware
//
#include <hip/hip_runtime.h>

typedef __attribute__((ext_vector_type(16))) _Float16 v16h;
typedef __attribute__((ext_vector_type(8)))  float    v8f;
typedef __attribute__((ext_vector_type(4)))  unsigned u32x4;
typedef __attribute__((ext_vector_type(8)))  int      i32x8;
typedef __attribute__((ext_vector_type(4)))  int      i32x4;

#define VOCABN 282
#define EMBEDN 7
#define HIDN   51
#define BATCHN 256
#define TLEN   1024
#define GP     208            // 4*HID=204 padded to 13 tiles of 16
#define CT1    13
#define K1T    2              // layer1 K = 64  -> 2 k-steps of 32
#define K2T    4              // layer2 K = 128 -> 4 k-steps of 32
#define VTN    18             // 282 padded to 288 -> 18 tiles
#define B64    (BATCHN * 64)

#define NWG    16             // independent scan workgroups (batch is row-parallel!)
#define ROWS   (BATCHN / NWG) // 16 rows per workgroup -> one 16-row WMMA tile
#define NTHR   256            // 8 waves per scan workgroup
#define NITER  ((ROWS * HIDN + NTHR - 1) / NTHR)   // 4 elementwise iterations

#define NW1 (K1T * CT1 * 32 * 16)   // 13312 halves
#define NW2 (K2T * CT1 * 32 * 16)   // 26624 halves
#define NWL (2   * VTN * 32 * 16)   // 18432 halves

__device__ __forceinline__ float sigf(float x) {
    return __builtin_amdgcn_rcpf(1.0f + __expf(-x));
}
__device__ __forceinline__ float tanh_fast(float x) {
    return 2.0f * __builtin_amdgcn_rcpf(1.0f + __expf(-2.0f * x)) - 1.0f;
}

// ---------------- weight pre-shuffle into WMMA B-fragment layout ----------------
// frag layout: W[((kt*NCT + ct)*32 + lane)*16 + i], element = Wlogical[k][c]
//   k = kt*32 + ((lane>>4)<<4) + i ; c = ct*16 + (lane & 15)
// biases folded into a weight row matched by a constant-1.0 activation column.

__global__ void k_prep_w1(const float* __restrict__ w_ih1, const float* __restrict__ w_hh1,
                          const float* __restrict__ b1, _Float16* __restrict__ W1f) {
    int idx = blockIdx.x * 256 + threadIdx.x;
    if (idx >= NW1) return;
    int i = idx & 15, lane = (idx >> 4) & 31, rem = idx >> 9;
    int ct = rem % CT1, kt = rem / CT1;
    int k = kt * 32 + ((lane >> 4) << 4) + i;
    int c = ct * 16 + (lane & 15);
    float v = 0.0f;
    if (c < 4 * HIDN) {
        if (k < EMBEDN)                  v = w_ih1[c * EMBEDN + k];        // x (cols 0..6)
        else if (k == EMBEDN)            v = b1[c];                        // bias row (col 7 == 1.0)
        else if (k >= 8 && k < 8 + HIDN) v = w_hh1[c * HIDN + (k - 8)];    // h1 (cols 8..58)
    }
    W1f[idx] = (_Float16)v;
}

__global__ void k_prep_w2(const float* __restrict__ w_ih2, const float* __restrict__ w_hh2,
                          const float* __restrict__ b2, _Float16* __restrict__ W2f) {
    int idx = blockIdx.x * 256 + threadIdx.x;
    if (idx >= NW2) return;
    int i = idx & 15, lane = (idx >> 4) & 31, rem = idx >> 9;
    int ct = rem % CT1, kt = rem / CT1;
    int k = kt * 32 + ((lane >> 4) << 4) + i;
    int c = ct * 16 + (lane & 15);
    float v = 0.0f;
    if (c < 4 * HIDN) {
        if (k < HIDN)                      v = w_ih2[c * HIDN + k];            // h1 (cols 0..50)
        else if (k == HIDN)                v = b2[c];                          // bias row (col 51 == 1.0)
        else if (k >= 64 && k < 64 + HIDN) v = w_hh2[c * HIDN + (k - 64)];     // h2 (cols 64..114)
    }
    W2f[idx] = (_Float16)v;
}

__global__ void k_prep_wl(const float* __restrict__ w_lin, const float* __restrict__ b_lin,
                          _Float16* __restrict__ Wlf) {
    int idx = blockIdx.x * 256 + threadIdx.x;
    if (idx >= NWL) return;
    int i = idx & 15, lane = (idx >> 4) & 31, rem = idx >> 9;
    int ct = rem % VTN, kt = rem / VTN;
    int k = kt * 32 + ((lane >> 4) << 4) + i;
    int c = ct * 16 + (lane & 15);
    float v = 0.0f;
    if (c < VOCABN) {
        if (k < HIDN)       v = w_lin[c * HIDN + k];
        else if (k == HIDN) v = b_lin[c];                                  // bias row (col 51 == 1.0)
    }
    Wlf[idx] = (_Float16)v;
}

// x_emb[t][b][0..7]: e<7 = embed, e==7 = 1.0 (bias partner column)
__global__ void k_prep_x(const int* __restrict__ input, const float* __restrict__ embed,
                         _Float16* __restrict__ xe) {
    int idx = blockIdx.x * 256 + threadIdx.x;
    if (idx >= TLEN * BATCHN * 8) return;
    int e = idx & 7, b = (idx >> 3) & (BATCHN - 1), t = idx >> 11;
    float v = 1.0f;
    if (e < EMBEDN) {
        int tok = input[b * TLEN + t];
        v = embed[tok * EMBEDN + e];
    }
    xe[idx] = (_Float16)v;
}

// h2_all bias partner column (col 51 = 1.0) for projection bias folding
__global__ void k_prep_h2pad(_Float16* __restrict__ h2all) {
    int idx = blockIdx.x * 256 + threadIdx.x;
    if (idx >= TLEN * BATCHN) return;
    h2all[(size_t)idx * 64 + HIDN] = (_Float16)1.0f;
}

// ---------------- sequential LSTM scan: 16 independent workgroups, 16 batch rows each ----
__global__ __launch_bounds__(NTHR, 1) void k_scan(
    const _Float16* __restrict__ xe, const _Float16* __restrict__ W1g,
    const _Float16* __restrict__ W2g, _Float16* __restrict__ h2all) {

    __shared__ _Float16 A1[ROWS][64];      // [x(0..6)|1.0|h1(8..58)|0]          2 KB
    __shared__ _Float16 H[ROWS][128];      // [h1(0..50)|1.0|0|h2(64..114)|0]    4 KB
    __shared__ _Float16 G[ROWS * GP];      // f16 gates for this slice          6.5 KB
    __shared__ _Float16 W1f[NW1];          //                                    26 KB
    __shared__ _Float16 W2f[NW2];          //                                    52 KB

    const int wg   = blockIdx.x;
    const int tid  = threadIdx.x;
    const int lane = tid & 31;
    const int wv   = tid >> 5;

    // ---- init: zero activations, stage weights into LDS ----
    for (int i = tid; i < ROWS * 64 / 2;  i += NTHR) ((unsigned*)A1)[i] = 0u;
    for (int i = tid; i < ROWS * 128 / 2; i += NTHR) ((unsigned*)H)[i]  = 0u;
    for (int i = tid; i < NW1 / 2; i += NTHR) ((unsigned*)W1f)[i] = ((const unsigned*)W1g)[i];
    for (int i = tid; i < NW2 / 2; i += NTHR) ((unsigned*)W2f)[i] = ((const unsigned*)W2g)[i];
    __syncthreads();
    if (tid < ROWS) H[tid][HIDN] = (_Float16)1.0f;   // layer2 bias partner

    // ---- per-wave WMMA tile slots (single 16-row tile; 13 col tiles over 8 waves) ----
    int nmy = 0;
    unsigned bOff[2];   // (ct*32 + lane)*16  (kt adds CT1*512)
    unsigned gOff[2];   // ((lane>>4)*8)*GP + ct*16 + (lane&15)
#pragma unroll
    for (int s = 0; s < 2; ++s) {
        int ct = wv + s * 8;
        if (ct < CT1) {
            bOff[s] = (unsigned)((ct * 32 + lane) * 16);
            gOff[s] = (unsigned)((((lane >> 4) << 3)) * GP + ct * 16 + (lane & 15));
            nmy = s + 1;
        }
    }
    const unsigned rowA = (unsigned)(lane & 15);
    const unsigned aK   = (unsigned)((lane >> 4) << 4);

    float c1r[NITER], c2r[NITER];
#pragma unroll
    for (int i = 0; i < NITER; ++i) { c1r[i] = 0.0f; c2r[i] = 0.0f; }

    // ---- TDM: per-step 2D tile, 16B rows x 16, LDS-padded to 128B row stride ----
    const unsigned ldsA1 = (unsigned)(unsigned long long)(&A1[0][0]);
    unsigned long long gbase = (unsigned long long)(const void*)xe
                             + (unsigned long long)wg * (ROWS * 8 * 2);
    // data_size=4B(2), pad_enable, pad_interval=1(4 dwords), pad_amount=27(28 dwords)
    i32x8 tg1 = { (int)((2u << 16) | (1u << 20) | (1u << 22) | (27u << 25)),
                  (int)(4u << 16),       // tensor_dim0 = 4
                  (int)(ROWS << 16),     // tensor_dim1 = 16
                  (int)(4u << 16),       // tile_dim0 = 4
                  (int)ROWS,             // tile_dim1 = 16
                  (int)4u,               // tensor_dim0_stride = 4
                  0, 0 };
    i32x4 tz4 = { 0, 0, 0, 0 };
    i32x8 tz8 = { 0, 0, 0, 0, 0, 0, 0, 0 };

    // preload x(0)
    if (wv == 0) {
        unsigned long long ga = gbase;
        u32x4 tg0 = { 1u, ldsA1, (unsigned)ga, (unsigned)(ga >> 32) | (2u << 30) };
        __builtin_amdgcn_tensor_load_to_lds(tg0, tg1, tz4, tz4, tz8, 0);
        __builtin_amdgcn_s_wait_tensorcnt(0);
    }
    __syncthreads();

    for (int t = 0; t < TLEN; ++t) {
        // ---- layer1 gates: G = A1 @ W1 (K=64, bias folded) ----
#pragma unroll
        for (int s = 0; s < 2; ++s) if (s < nmy) {
            const _Float16* ap = &A1[rowA][0];
            v8f acc = {};
#pragma unroll
            for (int kt = 0; kt < K1T; ++kt) {
                v16h a = *(const v16h*)(ap + kt * 32 + aK);
                v16h b = *(const v16h*)&W1f[bOff[s] + (unsigned)(kt * CT1 * 512)];
                acc = __builtin_amdgcn_wmma_f32_16x16x32_f16(false, a, false, b,
                                                             (short)0, acc, false, false);
            }
#pragma unroll
            for (int v = 0; v < 8; ++v)
                G[gOff[s] + (unsigned)(v * GP)] = (_Float16)acc[v];
        }
        __syncthreads();

        // x(t) consumed: DMA x(t+1) slice into A1 cols 0..7 (overlaps rest of step)
        if (wv == 0 && t + 1 < TLEN) {
            unsigned long long ga = gbase + (unsigned long long)(t + 1) * (BATCHN * 8 * 2);
            u32x4 tg0 = { 1u, ldsA1, (unsigned)ga, (unsigned)(ga >> 32) | (2u << 30) };
            __builtin_amdgcn_tensor_load_to_lds(tg0, tg1, tz4, tz4, tz8, 0);
        }

        // ---- layer1 elementwise ----
#pragma unroll
        for (int it = 0; it < NITER; ++it) {
            int idx = it * NTHR + tid;
            if (idx < ROWS * HIDN) {
                int bl = idx / HIDN, j = idx - bl * HIDN;
                int g0 = bl * GP + j;
                float ig = (float)G[g0];
                float fg = (float)G[g0 + HIDN];
                float gg = (float)G[g0 + 2 * HIDN];
                float og = (float)G[g0 + 3 * HIDN];
                float cn = sigf(fg) * c1r[it] + sigf(ig) * tanh_fast(gg);
                float h  = sigf(og) * tanh_fast(cn);
                c1r[it] = cn;
                _Float16 hh = (_Float16)h;
                A1[bl][8 + j] = hh;   // next step's layer1 input
                H[bl][j]      = hh;   // this step's layer2 input
            }
        }
        __syncthreads();

        // ---- layer2 gates: G = H @ W2 (K=128, bias folded) ----
#pragma unroll
        for (int s = 0; s < 2; ++s) if (s < nmy) {
            const _Float16* ap = &H[rowA][0];
            v8f acc = {};
#pragma unroll
            for (int kt = 0; kt < K2T; ++kt) {
                v16h a = *(const v16h*)(ap + kt * 32 + aK);
                v16h b = *(const v16h*)&W2f[bOff[s] + (unsigned)(kt * CT1 * 512)];
                acc = __builtin_amdgcn_wmma_f32_16x16x32_f16(false, a, false, b,
                                                             (short)0, acc, false, false);
            }
#pragma unroll
            for (int v = 0; v < 8; ++v)
                G[gOff[s] + (unsigned)(v * GP)] = (_Float16)acc[v];
        }
        __syncthreads();

        // ---- layer2 elementwise + h2 store ----
#pragma unroll
        for (int it = 0; it < NITER; ++it) {
            int idx = it * NTHR + tid;
            if (idx < ROWS * HIDN) {
                int bl = idx / HIDN, j = idx - bl * HIDN;
                int g0 = bl * GP + j;
                float ig = (float)G[g0];
                float fg = (float)G[g0 + HIDN];
                float gg = (float)G[g0 + 2 * HIDN];
                float og = (float)G[g0 + 3 * HIDN];
                float cn = sigf(fg) * c2r[it] + sigf(ig) * tanh_fast(gg);
                float h  = sigf(og) * tanh_fast(cn);
                c2r[it] = cn;
                _Float16 hh = (_Float16)h;
                H[bl][64 + j] = hh;
                h2all[(size_t)t * B64 + (size_t)(wg * ROWS + bl) * 64 + j] = hh;
            }
        }
        if (wv == 0) __builtin_amdgcn_s_wait_tensorcnt(0);  // x(t+1) landed
        __syncthreads();
    }
}

// ---------------- parallel vocab projection: out[b][t][v] = h2 @ w_lin.T + b_lin ----------------
__global__ __launch_bounds__(256) void k_proj(
    const _Float16* __restrict__ h2all, const _Float16* __restrict__ Wlf,
    float* __restrict__ out) {
    const int lane = threadIdx.x & 31;
    const int wv   = threadIdx.x >> 5;
    long gid = (long)blockIdx.x * 8 + wv;
    int vt = (int)(gid % VTN);
    long r  = gid / VTN;
    int tt = (int)(r % (TLEN / 16));
    int b  = (int)(r / (TLEN / 16));
    int t0 = tt * 16, v0 = vt * 16;

    v8f acc = {};
#pragma unroll
    for (int kt = 0; kt < 2; ++kt) {
        v16h a  = *(const v16h*)&h2all[(size_t)(t0 + (lane & 15)) * B64 +
                                       (size_t)b * 64 + kt * 32 + ((lane >> 4) << 4)];
        v16h bw = *(const v16h*)&Wlf[((kt * VTN + vt) * 32 + lane) * 16];
        acc = __builtin_amdgcn_wmma_f32_16x16x32_f16(false, a, false, bw,
                                                     (short)0, acc, false, false);
    }
    int n = v0 + (lane & 15);
#pragma unroll
    for (int v = 0; v < 8; ++v) {
        int m = v + ((lane >> 4) << 3);
        if (n < VOCABN)
            out[((size_t)b * TLEN + t0 + m) * VOCABN + n] = acc[v];
    }
}

extern "C" void kernel_launch(void* const* d_in, const int* in_sizes, int n_in,
                              void* d_out, int out_size, void* d_ws, size_t ws_size,
                              hipStream_t stream) {
    const int*   input = (const int*)d_in[0];
    const float* embed = (const float*)d_in[2];
    const float* w_ih1 = (const float*)d_in[3];
    const float* w_hh1 = (const float*)d_in[4];
    const float* b1    = (const float*)d_in[5];
    const float* w_ih2 = (const float*)d_in[6];
    const float* w_hh2 = (const float*)d_in[7];
    const float* b2    = (const float*)d_in[8];
    const float* w_lin = (const float*)d_in[9];
    const float* b_lin = (const float*)d_in[10];
    float* out = (float*)d_out;

    char* ws = (char*)d_ws;
    _Float16* W1g = (_Float16*)(ws + 0);              // 26,624 B
    _Float16* W2g = (_Float16*)(ws + (64 << 10));     // 53,248 B
    _Float16* Wlg = (_Float16*)(ws + (128 << 10));    // 36,864 B
    _Float16* xe  = (_Float16*)(ws + (256 << 10));    // 4 MB
    _Float16* h2  = (_Float16*)(ws + (8 << 20));      // 33.5 MB

    k_prep_w1<<<(NW1 + 255) / 256, 256, 0, stream>>>(w_ih1, w_hh1, b1, W1g);
    k_prep_w2<<<(NW2 + 255) / 256, 256, 0, stream>>>(w_ih2, w_hh2, b2, W2g);
    k_prep_wl<<<(NWL + 255) / 256, 256, 0, stream>>>(w_lin, b_lin, Wlg);
    k_prep_x<<<(TLEN * BATCHN * 8) / 256, 256, 0, stream>>>(input, embed, xe);
    k_prep_h2pad<<<(TLEN * BATCHN) / 256, 256, 0, stream>>>(h2);
    k_scan<<<NWG, NTHR, 0, stream>>>(xe, W1g, W2g, h2);
    k_proj<<<(BATCHN * (TLEN / 16) * VTN) / 8, 256, 0, stream>>>(h2, Wlg, out);
}
